// GATEncoder_31404800869119
// MI455X (gfx1250) — matrix operand order repaired and checked by reading, not compiled, not run on hardware
//
#include <hip/hip_runtime.h>
#include <hip/hip_bf16.h>

typedef _Float16 v4h  __attribute__((ext_vector_type(4)));
typedef _Float16 v8h  __attribute__((ext_vector_type(8)));
typedef _Float16 v16h __attribute__((ext_vector_type(16)));
typedef float    v8f  __attribute__((ext_vector_type(8)));
typedef int      v4i_vs __attribute__((vector_size(4 * sizeof(int))));

#define NEG_SLOPE 0.2f
#define BN_EPS 1e-5f

#if defined(__has_builtin)
#  if __has_builtin(__builtin_amdgcn_global_load_async_to_lds_b128) && \
      __has_builtin(__builtin_amdgcn_s_wait_asynccnt)
#    define USE_ASYNC_LDS 1
#  endif
#endif
#ifndef USE_ASYNC_LDS
#  define USE_ASYNC_LDS 0
#endif

__device__ __forceinline__ float lrelu(float x) { return x > 0.0f ? x : NEG_SLOPE * x; }

// float atomic max via monotonic int/uint encoding (lowers to global_atomic_max/min)
__device__ __forceinline__ void atomicMaxF(float* addr, float val) {
    if (val >= 0.0f) atomicMax((int*)addr, __float_as_int(val));
    else             atomicMin((unsigned int*)addr, __float_as_uint(val));
}

// 16-byte global -> LDS copy: async DMA path on gfx1250, sync fallback otherwise.
__device__ __forceinline__ void cp16(const _Float16* g, _Float16* l) {
#if USE_ASYNC_LDS
    __builtin_amdgcn_global_load_async_to_lds_b128(
        (__attribute__((address_space(1))) v4i_vs*)g,
        (__attribute__((address_space(3))) v4i_vs*)l, 0, 0);
#else
    *(float4*)l = *(const float4*)g;
#endif
}
__device__ __forceinline__ void cp16_wait() {
#if USE_ASYNC_LDS
    __builtin_amdgcn_s_wait_asynccnt(0);
#endif
}

// ---------------------------------------------------------------------------
// Pre-conversion kernels (run once per layer, not once per GEMM block)
// ---------------------------------------------------------------------------
__global__ void cvt_f16(const float* __restrict__ in, _Float16* __restrict__ out, int n4)
{
    int i = blockIdx.x * blockDim.x + threadIdx.x;
    if (i >= n4) return;
    float4 v = ((const float4*)in)[i];
    v4h h = { (_Float16)v.x, (_Float16)v.y, (_Float16)v.z, (_Float16)v.w };
    ((v4h*)out)[i] = h;
}

// Wt[n*128+k] = f16(W[k*128+n])  (128x128)
__global__ void prep_wt(const float* __restrict__ W, _Float16* __restrict__ Wt)
{
    int idx = blockIdx.x * blockDim.x + threadIdx.x;
    if (idx >= 128 * 128) return;
    int k = idx >> 7, n = idx & 127;
    Wt[n * 128 + k] = (_Float16)W[idx];
}

// ---------------------------------------------------------------------------
// WMMA GEMM: C[nrows,128] = A[nrows,128] @ B[128,128] (+ bias).
// A is pre-converted f16 row-major; B is pre-converted f16 transposed
// (Bt[n*128+k]) -- both already in the exact LDS layout, so staging is a
// pure byte copy (async-to-LDS). Block = 128 threads = 4 waves; each wave
// owns a 16-row stripe: 8 col tiles x 4 k-steps = 32 WMMA/wave.
// ---------------------------------------------------------------------------
__launch_bounds__(128)
__global__ void gemm_wmma_f16(const _Float16* __restrict__ Ah, const _Float16* __restrict__ Bt,
                              const float* __restrict__ bias, float* __restrict__ C, int nrows)
{
    __shared__ _Float16 sA[64 * 128];   // sA[m*128 + k], 16 KB
    __shared__ _Float16 sB[128 * 128];  // sB[n*128 + k], 32 KB

    const int t = threadIdx.x;              // 0..127
    const int rowBase = blockIdx.x * 64;

    // Stage B: 32 KB linear copy (16 x b128 per thread)
    #pragma unroll
    for (int i = 0; i < 16; ++i) {
        int h = i * 1024 + t * 8;
        cp16(Bt + h, sB + h);
    }
    // Stage A: 16 KB linear copy (8 x b128 per thread). Tail blocks may
    // over-read past row nrows (buffers placed so this stays in-bounds);
    // garbage rows are discarded by guarded stores.
    const _Float16* Ag = Ah + (size_t)rowBase * 128;
    #pragma unroll
    for (int i = 0; i < 8; ++i) {
        int h = i * 1024 + t * 8;
        cp16(Ag + h, sA + h);
    }
    cp16_wait();
    __syncthreads();

    const int wave   = t >> 5;
    const int lane   = t & 31;
    const int lane16 = lane & 15;
    const bool hiHalf = lane >= 16;
    const int mloc   = wave * 16;
    const int arow   = mloc + lane16;   // lanes 16-31 mirror M=0..15 with shifted K

    v8f acc[8] = {};

    #pragma unroll
    for (int s = 0; s < 4; ++s) {
        const int kb   = s * 32;
        // A fragment: 16-bit A 16x32 layout (lanes<16: K {kb..kb+7, kb+16..kb+23};
        // lanes>=16: K {kb+8..kb+15, kb+24..kb+31})
        const int aoff = kb + (hiHalf ? 8 : 0);
        const v8h* ap  = (const v8h*)(sA + arow * 128 + aoff);
        v8h alo = ap[0];
        v8h ahi = ap[2];   // +16 halves
        v16h afrag = __builtin_shufflevector(alo, ahi, 0,1,2,3,4,5,6,7,8,9,10,11,12,13,14,15);

        const int boff = kb + (hiHalf ? 8 : 0);
        #pragma unroll
        for (int n = 0; n < 8; ++n) {
            const int col = n * 16 + lane16;
            const v8h* bp = (const v8h*)(sB + col * 128 + boff);
            v8h blo = bp[0];
            v8h bhi = bp[2];
            v16h bfrag = __builtin_shufflevector(blo, bhi, 0,1,2,3,4,5,6,7,8,9,10,11,12,13,14,15);
            acc[n] = __builtin_amdgcn_wmma_f32_16x16x32_f16(
                false, afrag, false, bfrag, (short)0, acc[n], false, false);
        }
    }

    // f32 C/D layout: VGPR i -> (M=i, N=lane) lanes 0-15 ; (M=8+i, N=lane-16) lanes 16-31
    const int grow0 = rowBase + mloc + (hiHalf ? 8 : 0);
    #pragma unroll
    for (int n = 0; n < 8; ++n) {
        const int col = n * 16 + lane16;
        const float bv = bias ? bias[col] : 0.0f;
        #pragma unroll
        for (int i = 0; i < 8; ++i) {
            const int grow = grow0 + i;
            if (grow < nrows) C[(size_t)grow * 128 + col] = acc[n][i] + bv;
        }
    }
}

// ---------------------------------------------------------------------------
// Node kernel: es/ed per (node,head); init running max with self-loop value.
// ---------------------------------------------------------------------------
__global__ void attn_prep(const float* __restrict__ xW, const float* __restrict__ a_src,
                          const float* __restrict__ a_dst, float* __restrict__ es,
                          float* __restrict__ ed, float* __restrict__ m, int NH, int H, int Cdim)
{
    int idx = blockIdx.x * blockDim.x + threadIdx.x;
    if (idx >= NH) return;
    int n = idx / H, h = idx - n * H;
    const float* xr  = xW + (size_t)n * 128 + h * Cdim;
    const float* asr = a_src + h * Cdim;
    const float* adr = a_dst + h * Cdim;
    float s = 0.0f, d = 0.0f;
    for (int c = 0; c < Cdim; ++c) { float v = xr[c]; s += v * asr[c]; d += v * adr[c]; }
    es[idx] = s; ed[idx] = d;
    m[idx] = lrelu(s + d);   // self-loop edge (src==dst==n) seeds the max
}

__global__ void edge_max_k(const int* __restrict__ src, const int* __restrict__ dst,
                           const float* __restrict__ es, const float* __restrict__ ed,
                           float* __restrict__ m, int EH, int H)
{
    int idx = blockIdx.x * blockDim.x + threadIdx.x;
    if (idx >= EH) return;
    int e = idx / H, h = idx - e * H;
    int s = src[e], d = dst[e];
    float v = lrelu(es[s * H + h] + ed[d * H + h]);
    atomicMaxF(&m[d * H + h], v);
}

__global__ void denom_init(const float* __restrict__ es, const float* __restrict__ ed,
                           const float* __restrict__ m, float* __restrict__ den, int NH)
{
    int idx = blockIdx.x * blockDim.x + threadIdx.x;
    if (idx >= NH) return;
    float eL = lrelu(es[idx] + ed[idx]);
    den[idx] = expf(eL - m[idx]);   // self-loop contribution
}

__global__ void edge_sum_k(const int* __restrict__ src, const int* __restrict__ dst,
                           const float* __restrict__ es, const float* __restrict__ ed,
                           const float* __restrict__ m, float* __restrict__ den, int EH, int H)
{
    int idx = blockIdx.x * blockDim.x + threadIdx.x;
    if (idx >= EH) return;
    int e = idx / H, h = idx - e * H;
    int s = src[e], d = dst[e];
    float v = lrelu(es[s * H + h] + ed[d * H + h]);
    atomicAdd(&den[d * H + h], expf(v - m[d * H + h]));
}

// agg[n,f] = alpha_selfloop * xW[n,f]  (initializes output of segment_sum)
__global__ void agg_init(const float* __restrict__ xW, const float* __restrict__ es,
                         const float* __restrict__ ed, const float* __restrict__ m,
                         const float* __restrict__ den, float* __restrict__ agg,
                         int N128, int H, int cShift)
{
    int idx = blockIdx.x * blockDim.x + threadIdx.x;
    if (idx >= N128) return;
    int n = idx >> 7, f = idx & 127;
    int h = f >> cShift;
    float eL = lrelu(es[n * H + h] + ed[n * H + h]);
    float alpha = expf(eL - m[n * H + h]) / den[n * H + h];
    agg[idx] = alpha * xW[idx];
}

// One wave32 per edge: float4 gather of xW[src], 4 f32 atomic adds per lane.
__launch_bounds__(256)
__global__ void edge_agg(const int* __restrict__ src, const int* __restrict__ dst,
                         const float* __restrict__ xW, const float* __restrict__ es,
                         const float* __restrict__ ed, const float* __restrict__ m,
                         const float* __restrict__ den, float* __restrict__ agg,
                         int E, int H, int cShift)
{
    int gid  = blockIdx.x * blockDim.x + threadIdx.x;
    int eid  = gid >> 5;
    int lane = gid & 31;
    if (eid >= E) return;
    int s = src[eid], d = dst[eid];
    int f0 = lane * 4;
    int h  = f0 >> cShift;
    float e = lrelu(es[s * H + h] + ed[d * H + h]);
    float alpha = expf(e - m[d * H + h]) / den[d * H + h];
    float4 xv = *(const float4*)(xW + (size_t)s * 128 + f0);
    float* o = agg + (size_t)d * 128 + f0;
    atomicAdd(o + 0, alpha * xv.x);
    atomicAdd(o + 1, alpha * xv.y);
    atomicAdd(o + 2, alpha * xv.z);
    atomicAdd(o + 3, alpha * xv.w);
}

// out = agg + b + res ; optional BN(eval) + ReLU
__global__ void finalize_k(const float* __restrict__ agg, const float* __restrict__ b,
                           const float* __restrict__ res, const float* __restrict__ g,
                           const float* __restrict__ be, const float* __restrict__ rm,
                           const float* __restrict__ rv, int bnrelu,
                           float* __restrict__ dest, int N128)
{
    int idx = blockIdx.x * blockDim.x + threadIdx.x;
    if (idx >= N128) return;
    int f = idx & 127;
    float v = agg[idx] + b[f] + res[idx];
    if (bnrelu) {
        v = (v - rm[f]) * rsqrtf(rv[f] + BN_EPS) * g[f] + be[f];
        v = v > 0.0f ? v : 0.0f;
    }
    dest[idx] = v;
}

// ---------------------------------------------------------------------------
extern "C" void kernel_launch(void* const* d_in, const int* in_sizes, int n_in,
                              void* d_out, int out_size, void* d_ws, size_t ws_size,
                              hipStream_t stream)
{
    const float* x   = (const float*)d_in[0];
    const int*   ei  = (const int*)d_in[1];
    const float* W1  = (const float*)d_in[2];
    const float* a1s = (const float*)d_in[3];
    const float* a1d = (const float*)d_in[4];
    const float* b1  = (const float*)d_in[5];
    const float* W2  = (const float*)d_in[6];
    const float* a2s = (const float*)d_in[7];
    const float* a2d = (const float*)d_in[8];
    const float* b2  = (const float*)d_in[9];
    const float* W3  = (const float*)d_in[10];
    const float* a3s = (const float*)d_in[11];
    const float* a3d = (const float*)d_in[12];
    const float* b3  = (const float*)d_in[13];
    const float* R1w = (const float*)d_in[14];
    const float* R1b = (const float*)d_in[15];
    const float* R2w = (const float*)d_in[16];
    const float* R2b = (const float*)d_in[17];
    const float* R3w = (const float*)d_in[18];
    const float* R3b = (const float*)d_in[19];
    const float* g1  = (const float*)d_in[20];
    const float* be1 = (const float*)d_in[21];
    const float* rm1 = (const float*)d_in[22];
    const float* rv1 = (const float*)d_in[23];
    const float* g2  = (const float*)d_in[24];
    const float* be2 = (const float*)d_in[25];
    const float* rm2 = (const float*)d_in[26];
    const float* rv2 = (const float*)d_in[27];

    const int N = in_sizes[0] / 128;
    const int E = in_sizes[1] / 2;
    const int* src = ei;
    const int* dst = ei + E;

    // workspace layout (float-sized slots). Xh placed before Wt/Rt/es/... so
    // the GEMM's <16KB tail over-read stays inside the allocation.
    float* ws   = (float*)d_ws;
    size_t NF   = (size_t)N * 128;
    size_t NH8  = (size_t)N * 8;
    float*     xW   = ws;
    float*     res  = xW   + NF;
    float*     agg  = res  + NF;
    float*     hbuf = agg  + NF;
    _Float16*  Xh   = (_Float16*)(hbuf + NF);           // NF halves
    _Float16*  Wt   = (_Float16*)((float*)Xh + NF / 2); // 16384 halves
    _Float16*  Rt   = Wt + 16384;                       // 16384 halves
    float*     es   = (float*)(Rt + 16384);
    float*     ed   = es + NH8;
    float*     mm   = ed + NH8;
    float*     den  = mm + NH8;

    auto layer = [&](const float* X, const float* W, const float* asv, const float* adv,
                     const float* bv, const float* Rw, const float* Rb, int H,
                     const float* g, const float* be, const float* rm, const float* rv,
                     int bnrelu, float* dest)
    {
        const int Cdim   = 128 / H;
        const int cShift = (H == 8) ? 4 : 7;
        const int NH   = N * H;
        const int EH   = E * H;
        const int N128 = N * 128;
        const int N32  = N * 32;    // float4 chunks of X

        // once-per-layer pre-conversion (shared by both GEMMs / all blocks)
        cvt_f16<<<(N32 + 255) / 256, 256, 0, stream>>>(X, Xh, N32);
        prep_wt<<<64, 256, 0, stream>>>(W, Wt);
        prep_wt<<<64, 256, 0, stream>>>(Rw, Rt);

        dim3 gG((N + 63) / 64);
        gemm_wmma_f16<<<gG, 128, 0, stream>>>(Xh, Wt, nullptr, xW,  N);
        gemm_wmma_f16<<<gG, 128, 0, stream>>>(Xh, Rt, Rb,      res, N);

        attn_prep <<<(NH + 255) / 256, 256, 0, stream>>>(xW, asv, adv, es, ed, mm, NH, H, Cdim);
        edge_max_k<<<(EH + 255) / 256, 256, 0, stream>>>(src, dst, es, ed, mm, EH, H);
        denom_init<<<(NH + 255) / 256, 256, 0, stream>>>(es, ed, mm, den, NH);
        edge_sum_k<<<(EH + 255) / 256, 256, 0, stream>>>(src, dst, es, ed, mm, den, EH, H);
        agg_init  <<<(N128 + 255) / 256, 256, 0, stream>>>(xW, es, ed, mm, den, agg, N128, H, cShift);
        edge_agg  <<<(E + 7) / 8, 256, 0, stream>>>(src, dst, xW, es, ed, mm, den, agg, E, H, cShift);
        finalize_k<<<(N128 + 255) / 256, 256, 0, stream>>>(agg, bv, res, g, be, rm, rv, bnrelu, dest, N128);
    };

    layer(x,    W1, a1s, a1d, b1, R1w, R1b, 8, g1, be1, rm1, rv1, 1, hbuf);
    layer(hbuf, W2, a2s, a2d, b2, R2w, R2b, 8, g2, be2, rm2, rv2, 1, hbuf);
    layer(hbuf, W3, a3s, a3d, b3, R3w, R3b, 1, g1, be1, rm1, rv1, 0, (float*)d_out);
}
